// SwinBlock_67181878444336
// MI455X (gfx1250) — compile-verified, matrix operand order checked by reading, not run on hardware
//
#include <hip/hip_runtime.h>

// ---------------- constants ----------------
#define NWIN 2048      // 32 batches * 64 windows
#define NTOK 49        // 7*7
#define MPAD 64        // padded token rows per window (rows 49..63 are written as zeros)
#define CDIM 384
#define NHD  12
#define HIDD 1536

typedef unsigned int uint;
typedef uint  u32x4  __attribute__((ext_vector_type(4)));
typedef float f32x8  __attribute__((ext_vector_type(8)));
typedef __bf16 bf16x16 __attribute__((ext_vector_type(16)));

__device__ __forceinline__ unsigned short f2bf(float f) {
    union { float f; uint u; } v; v.f = f;
    uint r = v.u + 0x7fffu + ((v.u >> 16) & 1u);   // round-to-nearest-even
    return (unsigned short)(r >> 16);
}

__device__ __forceinline__ f32x8 wmma_bf(u32x4 al, u32x4 ah, u32x4 bl, u32x4 bh, f32x8 c) {
    union { u32x4 q[2]; bf16x16 v; } A, B;
    A.q[0] = al; A.q[1] = ah; B.q[0] = bl; B.q[1] = bh;
    // D = A(16x32 bf16) * B(32x16 bf16) + C(f32)
    return __builtin_amdgcn_wmma_f32_16x16x32_bf16(false, A.v, false, B.v, (short)0, c, false, false);
}

// Async stage of `nbytes` (multiple of 4096) from global to LDS by the whole 256-thread block.
// Uses CDNA5 async global->LDS DMA (ASYNCcnt) + s_wait_asynccnt, then workgroup barrier.
__device__ __forceinline__ void stage_async(const unsigned short* __restrict__ g,
                                            unsigned short* s, int n16 /* #16B chunks */) {
    uint ldsBase = (uint)(size_t)s;           // low 32 bits of flat LDS addr == LDS byte offset
    int  iters = n16 >> 8;                    // per-thread 16B chunks (n16 / 256)
    for (int i = 0; i < iters; ++i) {
        uint off = (uint)((i * 256 + threadIdx.x) << 4);
        uint lo  = ldsBase + off;
        asm volatile("global_load_async_to_lds_b128 %0, %1, %2"
                     :: "v"(lo), "v"(off), "s"(g) : "memory");
    }
    asm volatile("s_wait_asynccnt 0x0" ::: "memory");
    __syncthreads();
}

// ---------------- weight packing ----------------
// Pack W (OUT x IN, row-major f32) into per-(nt,kt)-tile B-fragment layout:
// dword index = tile*256 + lane*8 + d ; lane: n0=lane&15, half=lane>>4 ;
// d -> k = kt*32 + (d<4?0:16) + half*8 + (d&3)*2 ; dword = bf16(W[n,k+1])<<16 | bf16(W[n,k])
__global__ void k_pack(const float* __restrict__ W, int OUT, int IN, uint* __restrict__ dst) {
    int i = blockIdx.x * 256 + threadIdx.x;
    int total = OUT * IN / 2;
    if (i >= total) return;
    int t = i >> 8, lane = (i >> 3) & 31, d = i & 7;
    int KT = IN >> 5;
    int nt = t / KT, kt = t - nt * KT;
    int n0 = lane & 15, half = lane >> 4;
    int kk = kt * 32 + ((d & 4) << 2) + half * 8 + ((d & 3) << 1);
    const float* row = W + (size_t)(nt * 16 + n0) * IN + kk;
    dst[i] = ((uint)f2bf(row[1]) << 16) | (uint)f2bf(row[0]);
}

// bias_pad[h][m][n], 12x64x64 f32, -1e30 in padded region (self-masking softmax)
__global__ void k_bias(const float* __restrict__ table, const int* __restrict__ relidx,
                       float* __restrict__ biasPad) {
    int i = blockIdx.x * 256 + threadIdx.x;
    if (i >= 12 * 64 * 64) return;
    int h = i >> 12, m = (i >> 6) & 63, n = i & 63;
    float v = -1e30f;
    if (m < NTOK && n < NTOK) v = table[relidx[m * NTOK + n] * NHD + h];
    biasPad[i] = v;
}

// ---------------- layernorm (window partition on first pass); zero-fills pad rows ----------------
__global__ void k_ln(const float* __restrict__ src, int windowed,
                     const float* __restrict__ g, const float* __restrict__ b,
                     float* __restrict__ resid_out, unsigned short* __restrict__ a16) {
    int w = blockIdx.x;
    int lane = threadIdx.x & 31, wv = threadIdx.x >> 5;
    int bi = w >> 6, wi = w & 63, wr = wi >> 3, wc = wi & 7;
    for (int n = wv; n < MPAD; n += 8) {
        size_t o = ((size_t)w * MPAD + n) * CDIM;
        if (n >= NTOK) {            // zero pad row
            #pragma unroll
            for (int i = 0; i < 12; ++i) a16[o + lane + i * 32] = 0;
            continue;
        }
        const float* row;
        if (!windowed) {
            int r = n / 7, c = n - 7 * r;
            int l = (wr * 7 + r) * 56 + (wc * 7 + c);
            row = src + ((size_t)bi * 3136 + l) * CDIM;
        } else {
            row = src + ((size_t)w * NTOK + n) * CDIM;
        }
        float vals[12]; float s = 0.f;
        #pragma unroll
        for (int i = 0; i < 12; ++i) { vals[i] = row[lane + i * 32]; s += vals[i]; }
        for (int m = 16; m >= 1; m >>= 1) s += __shfl_xor(s, m, 32);
        float mean = s * (1.f / 384.f);
        float vs = 0.f;
        #pragma unroll
        for (int i = 0; i < 12; ++i) { float d = vals[i] - mean; vs += d * d; }
        for (int m = 16; m >= 1; m >>= 1) vs += __shfl_xor(vs, m, 32);
        float rstd = rsqrtf(vs * (1.f / 384.f) + 1e-5f);
        size_t ro = ((size_t)w * NTOK + n) * CDIM;
        #pragma unroll
        for (int i = 0; i < 12; ++i) {
            int c = lane + i * 32;
            float y = (vals[i] - mean) * rstd * g[c] + b[c];
            if (resid_out) resid_out[ro + c] = y;
            a16[o + c] = f2bf(y);
        }
    }
}

// ---------------- GEMM cores: one wave -> 64(M) x 16(N) strip, branch-free ----------------
// A staged in LDS (ld = CDIM), used by the K=384 GEMMs.
template<int KT>
__device__ __forceinline__ void gemm_core_lds(const unsigned short* sA,
                                              const uint* __restrict__ Bp, int nt,
                                              f32x8 acc[4], int lane) {
    const int m0 = lane & 15, half = lane >> 4;
    for (int kt = 0; kt < KT; ++kt) {
        const u32x4* bp = (const u32x4*)Bp + (size_t)(nt * KT + kt) * 64 + lane * 2;
        u32x4 bl = bp[0], bh = bp[1];
        __builtin_prefetch((const void*)(bp + 64), 0, 0);
        #pragma unroll
        for (int mt = 0; mt < 4; ++mt) {
            const unsigned short* ap = sA + (mt * 16 + m0) * CDIM + kt * 32 + half * 8;
            u32x4 al = *(const u32x4*)ap, ah = *(const u32x4*)(ap + 16);
            acc[mt] = wmma_bf(al, ah, bl, bh, acc[mt]);
        }
    }
}

// A direct from global (padded 64 rows), used by fc2 (A tile = 192KB > LDS budget).
template<int KT>
__device__ __forceinline__ void gemm_core_g(const unsigned short* __restrict__ A, int ld,
                                            const uint* __restrict__ Bp, int nt,
                                            f32x8 acc[4], int lane) {
    const int m0 = lane & 15, half = lane >> 4;
    for (int kt = 0; kt < KT; ++kt) {
        const u32x4* bp = (const u32x4*)Bp + (size_t)(nt * KT + kt) * 64 + lane * 2;
        u32x4 bl = bp[0], bh = bp[1];
        __builtin_prefetch((const void*)(bp + 64), 0, 0);
        #pragma unroll
        for (int mt = 0; mt < 4; ++mt) {
            const unsigned short* ap = A + (size_t)(mt * 16 + m0) * ld + kt * 32 + half * 8;
            u32x4 al = *(const u32x4*)ap, ah = *(const u32x4*)(ap + 16);
            acc[mt] = wmma_bf(al, ah, bl, bh, acc[mt]);
        }
    }
}

// ---------------- QKV GEMM: act(64x384,LDS) @ qkvW^T -> q,k rows + v transposed ----------------
__global__ void k_gemm_qkv(const unsigned short* __restrict__ act16, const uint* __restrict__ wq,
                           const float* __restrict__ qb,
                           unsigned short* __restrict__ qk16, unsigned short* __restrict__ vT) {
    __shared__ __align__(16) unsigned short sA[MPAD * CDIM];   // 48 KB
    int w = blockIdx.x;
    int lane = threadIdx.x & 31, wv = threadIdx.x >> 5;
    int nt = blockIdx.y * 8 + wv;                 // 0..71
    stage_async(act16 + (size_t)w * MPAD * CDIM, sA, MPAD * CDIM * 2 / 16);
    f32x8 acc[4];
    #pragma unroll
    for (int i = 0; i < 4; ++i) acc[i] = (f32x8)0.0f;
    gemm_core_lds<12>(sA, wq, nt, acc, lane);
    int n0 = lane & 15, half = lane >> 4;
    int col = nt * 16 + n0;
    float bb = qb[col];
    #pragma unroll
    for (int mt = 0; mt < 4; ++mt)
        #pragma unroll
        for (int r = 0; r < 8; ++r) {
            int tok = mt * 16 + half * 8 + r;
            unsigned short hv = (tok < NTOK) ? f2bf(acc[mt][r] + bb) : (unsigned short)0;
            if (col < 768) {
                qk16[((size_t)w * MPAD + tok) * 768 + col] = hv;
            } else {
                int hh = (col - 768) >> 5, dd = (col - 768) & 31;
                vT[(((size_t)w * NHD + hh) * 32 + dd) * 64 + tok] = hv;
            }
        }
}

// ---------------- attention: one head per wave; all frag loads branch-free ----------------
__global__ void k_attn(const unsigned short* __restrict__ qk16, const unsigned short* __restrict__ vT,
                       const float* __restrict__ biasPad, unsigned short* __restrict__ attn16) {
    __shared__ unsigned short probs[8 * NTOK * 72];   // 56,448 B; ld=72 keeps b128 ds aligned
    int lane = threadIdx.x & 31, wv = threadIdx.x >> 5;
    int p = blockIdx.x * 8 + wv;
    int win = p / NHD, h = p - win * NHD;
    int n0 = lane & 15, half = lane >> 4;
    const u32x4 z4 = {0, 0, 0, 0};
    const unsigned short* qkw = qk16 + (size_t)win * MPAD * 768;

    // Q A-fragments (K = head_dim = 32, exactly one wmma K-step); padded rows are zero
    u32x4 qlo[4], qhi[4];
    #pragma unroll
    for (int mt = 0; mt < 4; ++mt) {
        const unsigned short* ap = qkw + (size_t)(mt * 16 + n0) * 768 + h * 32 + half * 8;
        qlo[mt] = *(const u32x4*)ap; qhi[mt] = *(const u32x4*)(ap + 16);
    }
    // scores = Q @ K^T
    f32x8 acc[4][4];
    #pragma unroll
    for (int i = 0; i < 4; ++i)
        #pragma unroll
        for (int j = 0; j < 4; ++j) acc[i][j] = (f32x8)0.0f;
    for (int ntt = 0; ntt < 4; ++ntt) {
        const unsigned short* bp = qkw + (size_t)(ntt * 16 + n0) * 768 + 384 + h * 32 + half * 8;
        u32x4 bl = *(const u32x4*)bp, bh = *(const u32x4*)(bp + 16);
        #pragma unroll
        for (int mt = 0; mt < 4; ++mt)
            acc[mt][ntt] = wmma_bf(qlo[mt], qhi[mt], bl, bh, acc[mt][ntt]);
    }
    // bias + row softmax (rows live in 16-lane halves -> xor masks 8..1)
    const float scale = 0.17677669529663688f;   // 32^-0.5
    const float* bh2 = biasPad + (size_t)h * 4096;
    unsigned short* prbase = probs + wv * NTOK * 72;
    #pragma unroll
    for (int mt = 0; mt < 4; ++mt)
        #pragma unroll
        for (int r = 0; r < 8; ++r) {
            int m = mt * 16 + half * 8 + r;
            float s0 = acc[mt][0][r] * scale + bh2[m * 64 + 0  + n0];
            float s1 = acc[mt][1][r] * scale + bh2[m * 64 + 16 + n0];
            float s2 = acc[mt][2][r] * scale + bh2[m * 64 + 32 + n0];
            float s3 = acc[mt][3][r] * scale + bh2[m * 64 + 48 + n0];
            float mx = fmaxf(fmaxf(s0, s1), fmaxf(s2, s3));
            for (int mk = 8; mk >= 1; mk >>= 1) mx = fmaxf(mx, __shfl_xor(mx, mk, 32));
            float e0 = __expf(s0 - mx), e1 = __expf(s1 - mx);
            float e2 = __expf(s2 - mx), e3 = __expf(s3 - mx);
            float sum = e0 + e1 + e2 + e3;
            for (int mk = 8; mk >= 1; mk >>= 1) sum += __shfl_xor(sum, mk, 32);
            float inv = 1.f / sum;
            if (m < NTOK) {
                unsigned short* pr = prbase + m * 72;
                pr[0  + n0] = f2bf(e0 * inv);
                pr[16 + n0] = f2bf(e1 * inv);
                pr[32 + n0] = f2bf(e2 * inv);
                pr[48 + n0] = f2bf(e3 * inv);
            }
        }
    __syncthreads();
    // out = P(49x64) @ V(64x32)   (K padded to 64; padded probs are exactly 0)
    f32x8 oacc[4][2];
    #pragma unroll
    for (int i = 0; i < 4; ++i) { oacc[i][0] = (f32x8)0.0f; oacc[i][1] = (f32x8)0.0f; }
    for (int kt = 0; kt < 2; ++kt) {
        u32x4 vbl[2], vbh[2];
        #pragma unroll
        for (int nd = 0; nd < 2; ++nd) {
            int d = nd * 16 + n0;
            const unsigned short* vp = vT + (((size_t)win * NHD + h) * 32 + d) * 64
                                          + kt * 32 + half * 8;
            vbl[nd] = *(const u32x4*)vp; vbh[nd] = *(const u32x4*)(vp + 16);
        }
        #pragma unroll
        for (int mt = 0; mt < 4; ++mt) {
            int m = mt * 16 + n0;
            const unsigned short* ap = prbase + (size_t)(m < NTOK ? m : 0) * 72 + kt * 32 + half * 8;
            u32x4 al = z4, ah = z4;
            if (m < NTOK) { al = *(const u32x4*)ap; ah = *(const u32x4*)(ap + 16); }
            #pragma unroll
            for (int nd = 0; nd < 2; ++nd)
                oacc[mt][nd] = wmma_bf(al, ah, vbl[nd], vbh[nd], oacc[mt][nd]);
        }
    }
    #pragma unroll
    for (int mt = 0; mt < 4; ++mt)
        #pragma unroll
        for (int nd = 0; nd < 2; ++nd)
            #pragma unroll
            for (int r = 0; r < 8; ++r) {
                int tok = mt * 16 + half * 8 + r;
                unsigned short hv = (tok < NTOK) ? f2bf(oacc[mt][nd][r]) : (unsigned short)0;
                attn16[((size_t)win * MPAD + tok) * CDIM + h * 32 + nd * 16 + n0] = hv;
            }
}

// ---------------- proj GEMM (+residual into f32) ----------------
__global__ void k_gemm_proj(const unsigned short* __restrict__ attn16, const uint* __restrict__ wp,
                            const float* __restrict__ pb, float* __restrict__ resid) {
    __shared__ __align__(16) unsigned short sA[MPAD * CDIM];
    int w = blockIdx.x;
    int lane = threadIdx.x & 31, wv = threadIdx.x >> 5;
    int nt = blockIdx.y * 8 + wv;                 // 0..23
    stage_async(attn16 + (size_t)w * MPAD * CDIM, sA, MPAD * CDIM * 2 / 16);
    f32x8 acc[4];
    #pragma unroll
    for (int i = 0; i < 4; ++i) acc[i] = (f32x8)0.0f;
    gemm_core_lds<12>(sA, wp, nt, acc, lane);
    int n0 = lane & 15, half = lane >> 4;
    int col = nt * 16 + n0;
    float bb = pb[col];
    #pragma unroll
    for (int mt = 0; mt < 4; ++mt)
        #pragma unroll
        for (int r = 0; r < 8; ++r) {
            int tok = mt * 16 + half * 8 + r;
            if (tok < NTOK) {
                size_t o = ((size_t)w * NTOK + tok) * CDIM + col;
                resid[o] = resid[o] + acc[mt][r] + bb;
            }
        }
}

// ---------------- fc1 GEMM + exact GELU ----------------
__global__ void k_gemm_fc1(const unsigned short* __restrict__ act16, const uint* __restrict__ wf,
                           const float* __restrict__ fb, unsigned short* __restrict__ hmid) {
    __shared__ __align__(16) unsigned short sA[MPAD * CDIM];
    int w = blockIdx.x;
    int lane = threadIdx.x & 31, wv = threadIdx.x >> 5;
    int nt = blockIdx.y * 8 + wv;                 // 0..95
    stage_async(act16 + (size_t)w * MPAD * CDIM, sA, MPAD * CDIM * 2 / 16);
    f32x8 acc[4];
    #pragma unroll
    for (int i = 0; i < 4; ++i) acc[i] = (f32x8)0.0f;
    gemm_core_lds<12>(sA, wf, nt, acc, lane);
    int n0 = lane & 15, half = lane >> 4;
    int col = nt * 16 + n0;
    float bb = fb[col];
    #pragma unroll
    for (int mt = 0; mt < 4; ++mt)
        #pragma unroll
        for (int r = 0; r < 8; ++r) {
            int tok = mt * 16 + half * 8 + r;
            unsigned short hv = 0;
            if (tok < NTOK) {
                float v = acc[mt][r] + bb;
                hv = f2bf(0.5f * v * (1.0f + erff(v * 0.70710678118654752f)));
            }
            hmid[((size_t)w * MPAD + tok) * HIDD + col] = hv;
        }
}

// ---------------- fc2 GEMM + residual + window reverse -> d_out ----------------
__global__ void k_gemm_fc2(const unsigned short* __restrict__ hmid, const uint* __restrict__ wf,
                           const float* __restrict__ fb, const float* __restrict__ resid,
                           float* __restrict__ out) {
    int w = blockIdx.x;
    int lane = threadIdx.x & 31, wv = threadIdx.x >> 5;
    int nt = blockIdx.y * 8 + wv;                 // 0..23
    f32x8 acc[4];
    #pragma unroll
    for (int i = 0; i < 4; ++i) acc[i] = (f32x8)0.0f;
    gemm_core_g<48>(hmid + (size_t)w * MPAD * HIDD, HIDD, wf, nt, acc, lane);
    int n0 = lane & 15, half = lane >> 4;
    int col = nt * 16 + n0;
    float bb = fb[col];
    int bi = w >> 6, wi = w & 63, wr = wi >> 3, wc = wi & 7;
    #pragma unroll
    for (int mt = 0; mt < 4; ++mt)
        #pragma unroll
        for (int r = 0; r < 8; ++r) {
            int tok = mt * 16 + half * 8 + r;
            if (tok < NTOK) {
                float v = acc[mt][r] + bb + resid[((size_t)w * NTOK + tok) * CDIM + col];
                int rr = tok / 7, cc = tok - 7 * rr;
                int l = (wr * 7 + rr) * 56 + (wc * 7 + cc);
                out[((size_t)bi * 3136 + l) * CDIM + col] = v;
            }
        }
}

// ---------------- host launcher ----------------
extern "C" void kernel_launch(void* const* d_in, const int* in_sizes, int n_in,
                              void* d_out, int out_size, void* d_ws, size_t ws_size,
                              hipStream_t stream) {
    const float* x     = (const float*)d_in[0];
    const float* n1g   = (const float*)d_in[1];
    const float* n1b   = (const float*)d_in[2];
    const float* qkvW  = (const float*)d_in[3];
    const float* qkvB  = (const float*)d_in[4];
    const float* relT  = (const float*)d_in[5];
    const int*   relI  = (const int*)d_in[6];
    const float* projW = (const float*)d_in[7];
    const float* projB = (const float*)d_in[8];
    const float* n2g   = (const float*)d_in[9];
    const float* n2b   = (const float*)d_in[10];
    const float* fc1W  = (const float*)d_in[11];
    const float* fc1B  = (const float*)d_in[12];
    const float* fc2W  = (const float*)d_in[13];
    const float* fc2B  = (const float*)d_in[14];
    float* out = (float*)d_out;
    char* ws = (char*)d_ws;

    constexpr size_t R64 = (size_t)NWIN * MPAD;   // 131072 padded rows
    size_t off = 0;
    float* resid = (float*)(ws + off);                    off += (size_t)NWIN * NTOK * CDIM * 4;
    unsigned short* act16  = (unsigned short*)(ws + off); off += R64 * CDIM * 2;
    size_t qOff = off;
    unsigned short* qk16   = (unsigned short*)(ws + off); off += R64 * 768 * 2;
    unsigned short* vT     = (unsigned short*)(ws + off); off += (size_t)NWIN * NHD * 32 * 64 * 2;
    unsigned short* attn16 = (unsigned short*)(ws + off); off += R64 * CDIM * 2;
    // fc1 output overlaps qk16+vT+attn16 (all dead once fc1 runs): R64*1536*2 bytes exactly
    unsigned short* hmid16 = (unsigned short*)(ws + qOff);
    size_t woff = off;
    uint* pwQ  = (uint*)(ws + woff); woff += (size_t)1152 * 384 * 2;
    uint* pwP  = (uint*)(ws + woff); woff += (size_t)384  * 384 * 2;
    uint* pwF1 = (uint*)(ws + woff); woff += (size_t)1536 * 384 * 2;
    uint* pwF2 = (uint*)(ws + woff); woff += (size_t)384  * 1536 * 2;
    float* biasPad = (float*)(ws + woff);

    dim3 B256(256);
    k_pack<<<dim3((1152 * 384 / 2 + 255) / 256), B256, 0, stream>>>(qkvW, 1152, 384, pwQ);
    k_pack<<<dim3((384  * 384 / 2 + 255) / 256), B256, 0, stream>>>(projW, 384, 384, pwP);
    k_pack<<<dim3((1536 * 384 / 2 + 255) / 256), B256, 0, stream>>>(fc1W, 1536, 384, pwF1);
    k_pack<<<dim3((384 * 1536 / 2 + 255) / 256), B256, 0, stream>>>(fc2W, 384, 1536, pwF2);
    k_bias<<<dim3(192), B256, 0, stream>>>(relT, relI, biasPad);

    k_ln<<<dim3(NWIN), B256, 0, stream>>>(x, 0, n1g, n1b, resid, act16);
    k_gemm_qkv<<<dim3(NWIN, 9), B256, 0, stream>>>(act16, pwQ, qkvB, qk16, vT);
    k_attn<<<dim3(NWIN * NHD / 8), B256, 0, stream>>>(qk16, vT, biasPad, attn16);
    k_gemm_proj<<<dim3(NWIN, 3), B256, 0, stream>>>(attn16, pwP, projB, resid);
    k_ln<<<dim3(NWIN), B256, 0, stream>>>(resid, 1, n2g, n2b, nullptr, act16);
    k_gemm_fc1<<<dim3(NWIN, 12), B256, 0, stream>>>(act16, pwF1, fc1B, hmid16);
    k_gemm_fc2<<<dim3(NWIN, 3), B256, 0, stream>>>(hmid16, pwF2, fc2B, resid, out);
}
